// FraudGNN_6622839570533
// MI455X (gfx1250) — matrix operand order repaired and checked by reading, not compiled, not run on hardware
//
#include <hip/hip_runtime.h>
#include <hip/hip_bf16.h>

typedef __attribute__((ext_vector_type(16))) _Float16 v16h;
typedef __attribute__((ext_vector_type(8)))  _Float16 v8h;
typedef __attribute__((ext_vector_type(8)))  float    v8f;

// ---------------------------------------------------------------------------
// Pack stacked weights [Wl; Wr] (row-major [64, K*]) into the exact per-lane
// B-operand layout for v_wmma_f32_16x16x32_f16:
//   lane L holds column N = L&15; half j holds K = ks*32 + (L>>4)*8 + (j&7) + (j>>3)*16.
// Output index: ((ct*KS + ks)*32 + lane)*16 + j   (4 column tiles of 16).
// ---------------------------------------------------------------------------
__global__ __launch_bounds__(256) void pack_b_kernel(
    const float* __restrict__ Wl, const float* __restrict__ Wr,
    int Kl, int Kr, int KS, _Float16* __restrict__ bpack)
{
    int idx = blockIdx.x * 256 + threadIdx.x;
    int total = 4 * KS * 32 * 16;
    if (idx >= total) return;
    int j    = idx & 15;
    int lane = (idx >> 4) & 31;
    int rest = idx >> 9;            // ct*KS + ks
    int ks   = rest % KS;
    int ct   = rest / KS;
    int K    = ks * 32 + (lane >> 4) * 8 + (j & 7) + (j >> 3) * 16;
    int c    = ct * 16 + (lane & 15);
    float v = 0.f;
    if (K < Kl)            v = Wl[c * Kl + K];
    else if (K < Kl + Kr)  v = Wr[c * Kr + (K - Kl)];
    bpack[idx] = (_Float16)v;
}

// ---------------------------------------------------------------------------
// Edge pass 1: one wave32 per edge.
//   - lane c / c+32: enc = relu(We@eattr + be)  -> atomic add into hacc[dst]
//   - lanes 0..20:   aggx[dst][f] += x[src][f]
//   - lane 31:       deg[dst] += 1
// ---------------------------------------------------------------------------
__global__ __launch_bounds__(256) void edge1_kernel(
    const int* __restrict__ ei, const float* __restrict__ eattr,
    const float* __restrict__ x,
    const float* __restrict__ We, const float* __restrict__ be,
    float* __restrict__ aggx, float* __restrict__ hacc,
    float* __restrict__ deg, int E)
{
    __shared__ float sWe[192];
    __shared__ float sbe[64];
    int tid = threadIdx.x;
    if (tid < 192) sWe[tid] = We[tid];
    else           sbe[tid - 192] = be[tid - 192];
    __syncthreads();

    int gw   = (blockIdx.x * 256 + tid) >> 5;
    int lane = tid & 31;
    if (gw >= E) return;

    int src = ei[gw];
    int dst = ei[(size_t)E + gw];
    float a0 = eattr[(size_t)gw * 3 + 0];
    float a1 = eattr[(size_t)gw * 3 + 1];
    float a2 = eattr[(size_t)gw * 3 + 2];

#pragma unroll
    for (int t = 0; t < 2; ++t) {
        int c = lane + t * 32;
        float enc = fmaxf(sWe[c*3]*a0 + sWe[c*3+1]*a1 + sWe[c*3+2]*a2 + sbe[c], 0.f);
        atomicAdd(&hacc[(size_t)dst * 64 + c], enc);
    }
    if (lane < 21) atomicAdd(&aggx[(size_t)dst * 24 + lane], x[(size_t)src * 21 + lane]);
    if (lane == 31) atomicAdd(&deg[dst], 1.0f);
}

// ---------------------------------------------------------------------------
// Edge pass 2: one wave32 per edge; lane handles channels {2l, 2l+1}.
// Applies BN1 scale/shift + relu on the fly while gathering h1pre[src].
// ---------------------------------------------------------------------------
__global__ __launch_bounds__(256) void edge2_kernel(
    const int* __restrict__ ei, const float* __restrict__ h1pre,
    const float* __restrict__ scsh1, float* __restrict__ agg2, int E)
{
    __shared__ float sSc[64];
    __shared__ float sSh[64];
    int tid = threadIdx.x;
    if (tid < 64)       sSc[tid] = scsh1[tid];
    else if (tid < 128) sSh[tid - 64] = scsh1[tid];
    __syncthreads();

    int gw   = (blockIdx.x * 256 + tid) >> 5;
    int lane = tid & 31;
    if (gw >= E) return;

    int src = ei[gw];
    int dst = ei[(size_t)E + gw];
    int c0 = lane * 2, c1 = c0 + 1;
    float2 v = *(const float2*)&h1pre[(size_t)src * 64 + c0];
    float y0 = fmaxf(v.x * sSc[c0] + sSh[c0], 0.f);
    float y1 = fmaxf(v.y * sSc[c1] + sSh[c1], 0.f);
    atomicAdd(&agg2[(size_t)dst * 64 + c0], y0);
    atomicAdd(&agg2[(size_t)dst * 64 + c1], y1);
}

// ---------------------------------------------------------------------------
// Node pass 1: per-wave 16-node x 64-ch tile.
//   A = [agg*inv_deg (21) | x (21) | 0-pad] as f16 16x64, staged in LDS.
//   B = packed [W1l;W1r].  D += b1l + edge_agg.  BN1 stats via LDS atomics.
// ---------------------------------------------------------------------------
__global__ __launch_bounds__(256) void node1_kernel(
    const float* __restrict__ x, const float* __restrict__ aggx,
    const float* __restrict__ deg, const float* __restrict__ hacc,
    const float* __restrict__ b1l, const _Float16* __restrict__ bpack,
    float* __restrict__ h1pre, float* __restrict__ stats, int N)
{
    __shared__ __align__(16) _Float16 Asmem[8][16 * 64];
    __shared__ float sSum[64];
    __shared__ float sSq[64];
    int tid = threadIdx.x;
    if (tid < 64) { sSum[tid] = 0.f; sSq[tid] = 0.f; }
    __syncthreads();

    int wave = tid >> 5, lane = tid & 31;
    int tileBase = (blockIdx.x * 8 + wave) * 16;
    _Float16* A = Asmem[wave];

    for (int idx = lane; idx < 16 * 64; idx += 32) {
        int m = idx >> 6, k = idx & 63;
        int n = tileBase + m;
        float v = 0.f;
        if (n < N) {
            if (k < 21)      v = aggx[(size_t)n * 24 + k] * (1.0f / fmaxf(deg[n], 1.0f));
            else if (k < 42) v = x[(size_t)n * 21 + (k - 21)];
        }
        A[idx] = (_Float16)v;
    }
    __syncthreads();

    const int M = lane & 15, hi = lane >> 4;
    const bool full = (tileBase + 16 <= N);   // wave-uniform fast path
    for (int ct = 0; ct < 4; ++ct) {
        v8f acc = {};
#pragma unroll
        for (int ks = 0; ks < 2; ++ks) {
            int base = hi * 8 + ks * 32;
            v8h lo = *(const v8h*)&A[M * 64 + base];
            v8h hh = *(const v8h*)&A[M * 64 + base + 16];
            v16h a;
#pragma unroll
            for (int i = 0; i < 8; ++i) { a[i] = lo[i]; a[8 + i] = hh[i]; }
            v16h b = *(const v16h*)&bpack[(size_t)((ct * 2 + ks) * 32 + lane) * 16];
            acc = __builtin_amdgcn_wmma_f32_16x16x32_f16(false, a, false, b,
                                                         (short)0, acc, false, false);
        }
        const int ch = ct * 16 + M;
        const float bias = b1l[ch];
        float s = 0.f, s2 = 0.f;
        if (full) {
            const size_t rowBase = (size_t)(tileBase + hi * 8) * 64 + ch;
            float hv[8];
#pragma unroll
            for (int r = 0; r < 8; ++r) hv[r] = hacc[rowBase + (size_t)r * 64];
#pragma unroll
            for (int r = 0; r < 8; ++r) {
                float v = acc[r] + bias + hv[r];
                h1pre[rowBase + (size_t)r * 64] = v;
                s += v; s2 += v * v;
            }
        } else {
#pragma unroll
            for (int r = 0; r < 8; ++r) {
                int n = tileBase + hi * 8 + r;
                if (n < N) {
                    float v = acc[r] + bias + hacc[(size_t)n * 64 + ch];
                    h1pre[(size_t)n * 64 + ch] = v;
                    s += v; s2 += v * v;
                }
            }
        }
        atomicAdd(&sSum[ch], s);
        atomicAdd(&sSq[ch], s2);
    }
    __syncthreads();
    if (tid < 64) {
        atomicAdd(&stats[tid], sSum[tid]);
        atomicAdd(&stats[64 + tid], sSq[tid]);
    }
}

// ---------------------------------------------------------------------------
// Node pass 2: K=128.  A = [agg2*inv_deg (64) | relu(bn1(h1pre)) (64)].
// ---------------------------------------------------------------------------
__global__ __launch_bounds__(256) void node2_kernel(
    const float* __restrict__ h1pre, const float* __restrict__ agg2,
    const float* __restrict__ deg, const float* __restrict__ scsh1,
    const float* __restrict__ b2l, const _Float16* __restrict__ bpack,
    float* __restrict__ h2pre, float* __restrict__ stats, int N)
{
    __shared__ __align__(16) _Float16 Asmem[8][16 * 128];
    __shared__ float sSum[64];
    __shared__ float sSq[64];
    __shared__ float sSc[64];
    __shared__ float sSh[64];
    int tid = threadIdx.x;
    if (tid < 64) {
        sSum[tid] = 0.f; sSq[tid] = 0.f;
        sSc[tid] = scsh1[tid]; sSh[tid] = scsh1[64 + tid];
    }
    __syncthreads();

    int wave = tid >> 5, lane = tid & 31;
    int tileBase = (blockIdx.x * 8 + wave) * 16;
    _Float16* A = Asmem[wave];

    for (int idx = lane; idx < 16 * 128; idx += 32) {
        int m = idx >> 7, k = idx & 127;
        int n = tileBase + m;
        float v = 0.f;
        if (n < N) {
            if (k < 64) v = agg2[(size_t)n * 64 + k] * (1.0f / fmaxf(deg[n], 1.0f));
            else {
                int c = k - 64;
                v = fmaxf(h1pre[(size_t)n * 64 + c] * sSc[c] + sSh[c], 0.f);
            }
        }
        A[idx] = (_Float16)v;
    }
    __syncthreads();

    const int M = lane & 15, hi = lane >> 4;
    const bool full = (tileBase + 16 <= N);   // wave-uniform fast path
    for (int ct = 0; ct < 4; ++ct) {
        v8f acc = {};
#pragma unroll
        for (int ks = 0; ks < 4; ++ks) {
            int base = hi * 8 + ks * 32;
            v8h lo = *(const v8h*)&A[M * 128 + base];
            v8h hh = *(const v8h*)&A[M * 128 + base + 16];
            v16h a;
#pragma unroll
            for (int i = 0; i < 8; ++i) { a[i] = lo[i]; a[8 + i] = hh[i]; }
            v16h b = *(const v16h*)&bpack[(size_t)((ct * 4 + ks) * 32 + lane) * 16];
            acc = __builtin_amdgcn_wmma_f32_16x16x32_f16(false, a, false, b,
                                                         (short)0, acc, false, false);
        }
        const int ch = ct * 16 + M;
        const float bias = b2l[ch];
        float s = 0.f, s2 = 0.f;
        if (full) {
            const size_t rowBase = (size_t)(tileBase + hi * 8) * 64 + ch;
#pragma unroll
            for (int r = 0; r < 8; ++r) {
                float v = acc[r] + bias;
                h2pre[rowBase + (size_t)r * 64] = v;
                s += v; s2 += v * v;
            }
        } else {
#pragma unroll
            for (int r = 0; r < 8; ++r) {
                int n = tileBase + hi * 8 + r;
                if (n < N) {
                    float v = acc[r] + bias;
                    h2pre[(size_t)n * 64 + ch] = v;
                    s += v; s2 += v * v;
                }
            }
        }
        atomicAdd(&sSum[ch], s);
        atomicAdd(&sSq[ch], s2);
    }
    __syncthreads();
    if (tid < 64) {
        atomicAdd(&stats[tid], sSum[tid]);
        atomicAdd(&stats[64 + tid], sSq[tid]);
    }
}

// BN finalize: scale = g*rsqrt(var+eps), shift = beta - mu*scale (biased var).
__global__ void bnfin_kernel(const float* __restrict__ stats,
                             const float* __restrict__ g, const float* __restrict__ beta,
                             float* __restrict__ scsh, float n_inv)
{
    int c = threadIdx.x;
    if (c < 64) {
        float mu  = stats[c] * n_inv;
        float var = stats[64 + c] * n_inv - mu * mu;
        float s   = g[c] * rsqrtf(var + 1e-5f);
        scsh[c]      = s;
        scsh[64 + c] = beta[c] - mu * s;
    }
}

// Classifier: out = relu(bn2(h2pre)) @ Wc.T + bc  ([N,2])
__global__ __launch_bounds__(256) void classifier_kernel(
    const float* __restrict__ h2pre, const float* __restrict__ scsh2,
    const float* __restrict__ Wc, const float* __restrict__ bc,
    float* __restrict__ out, int N)
{
    __shared__ float sW[128];
    __shared__ float sSc[64];
    __shared__ float sSh[64];
    __shared__ float sB[2];
    int tid = threadIdx.x;
    if (tid < 128) sW[tid] = Wc[tid];
    if (tid < 64) { sSc[tid] = scsh2[tid]; sSh[tid] = scsh2[64 + tid]; }
    if (tid < 2)  sB[tid] = bc[tid];
    __syncthreads();

    int n = blockIdx.x * 256 + tid;
    if (n >= N) return;
    const float4* h4 = (const float4*)&h2pre[(size_t)n * 64];
    float a0 = sB[0], a1 = sB[1];
#pragma unroll
    for (int q = 0; q < 16; ++q) {
        float4 hv = h4[q];
        int k = q * 4;
        float v0 = fmaxf(hv.x * sSc[k+0] + sSh[k+0], 0.f);
        float v1 = fmaxf(hv.y * sSc[k+1] + sSh[k+1], 0.f);
        float v2 = fmaxf(hv.z * sSc[k+2] + sSh[k+2], 0.f);
        float v3 = fmaxf(hv.w * sSc[k+3] + sSh[k+3], 0.f);
        a0 += v0*sW[k+0] + v1*sW[k+1] + v2*sW[k+2] + v3*sW[k+3];
        a1 += v0*sW[64+k+0] + v1*sW[64+k+1] + v2*sW[64+k+2] + v3*sW[64+k+3];
    }
    out[(size_t)n * 2 + 0] = a0;
    out[(size_t)n * 2 + 1] = a1;
}

extern "C" void kernel_launch(void* const* d_in, const int* in_sizes, int n_in,
                              void* d_out, int out_size, void* d_ws, size_t ws_size,
                              hipStream_t stream)
{
    const float* x   = (const float*)d_in[0];
    const int*   ei  = (const int*)  d_in[1];
    const float* ea  = (const float*)d_in[2];
    const float* We  = (const float*)d_in[3];
    const float* be  = (const float*)d_in[4];
    const float* W1l = (const float*)d_in[5];
    const float* b1l = (const float*)d_in[6];
    const float* W1r = (const float*)d_in[7];
    const float* W2l = (const float*)d_in[8];
    const float* b2l = (const float*)d_in[9];
    const float* W2r = (const float*)d_in[10];
    const float* g1  = (const float*)d_in[11];
    const float* bt1 = (const float*)d_in[12];
    const float* g2  = (const float*)d_in[13];
    const float* bt2 = (const float*)d_in[14];
    const float* Wc  = (const float*)d_in[15];
    const float* bc  = (const float*)d_in[16];

    const int N = in_sizes[0] / 21;
    const int E = in_sizes[1] / 2;
    const size_t N_ = (size_t)N;

    // Workspace layout (float offsets, 32B-aligned regions).
    float* w = (float*)d_ws;
    size_t f = 0;
    size_t o_deg  = f; f += N_;            f = (f + 7) & ~(size_t)7;
    size_t o_st1  = f; f += 128;
    size_t o_st2  = f; f += 128;
    size_t o_sc1  = f; f += 128;
    size_t o_sc2  = f; f += 128;
    size_t o_bp1  = f; f += 2048;          // 4096 f16
    size_t o_bp2  = f; f += 4096;          // 8192 f16
    size_t o_aggx = f; f += 24 * N_;
    size_t o_acc  = f; f += 64 * N_;       // h1acc, reused as agg2
    size_t zeroEnd = f;
    size_t o_h1   = f; f += 64 * N_;
    size_t o_h2   = f; f += 64 * N_;
    (void)ws_size; (void)n_in; (void)out_size;

    _Float16* bp1 = (_Float16*)(w + o_bp1);
    _Float16* bp2 = (_Float16*)(w + o_bp2);

    // Zero all accumulators (deg, stats, aggx, edge/h accumulator).
    hipMemsetAsync(w, 0, zeroEnd * sizeof(float), stream);

    // Pack WMMA B operands.
    pack_b_kernel<<<16, 256, 0, stream>>>(W1l, W1r, 21, 21, 2, bp1);
    pack_b_kernel<<<32, 256, 0, stream>>>(W2l, W2r, 64, 64, 4, bp2);

    const int eb = (E + 7) / 8;     // 8 wave32 per block, one edge per wave
    const int nb = (N + 127) / 128; // 8 tiles of 16 nodes per block

    // Layer 1
    edge1_kernel<<<eb, 256, 0, stream>>>(ei, ea, x, We, be,
                                         w + o_aggx, w + o_acc, w + o_deg, E);
    node1_kernel<<<nb, 256, 0, stream>>>(x, w + o_aggx, w + o_deg, w + o_acc,
                                         b1l, bp1, w + o_h1, w + o_st1, N);
    bnfin_kernel<<<1, 64, 0, stream>>>(w + o_st1, g1, bt1, w + o_sc1, 1.0f / (float)N);

    // Layer 2 (reuse o_acc as agg2 accumulator)
    hipMemsetAsync(w + o_acc, 0, 64 * N_ * sizeof(float), stream);
    edge2_kernel<<<eb, 256, 0, stream>>>(ei, w + o_h1, w + o_sc1, w + o_acc, E);
    node2_kernel<<<nb, 256, 0, stream>>>(w + o_h1, w + o_acc, w + o_deg, w + o_sc1,
                                         b2l, bp2, w + o_h2, w + o_st2, N);
    bnfin_kernel<<<1, 64, 0, stream>>>(w + o_st2, g2, bt2, w + o_sc2, 1.0f / (float)N);

    // Classifier
    classifier_kernel<<<(N + 255) / 256, 256, 0, stream>>>(
        w + o_h2, w + o_sc2, Wc, bc, (float*)d_out, N);
}
